// miniEncoder_51926154608961
// MI455X (gfx1250) — matrix operand order repaired and checked by reading, not compiled, not run on hardware
//
#include <hip/hip_runtime.h>
#include <hip/hip_bf16.h>

// ---------------------------------------------------------------------------
// miniEncoder for MI455X (gfx1250).
// bf16 WMMA GEMMs (fp32 accum), bf16 activation pipeline, async LDS staging.
// Shapes: B=4, S=96, T=96, D=512, H=8, DK=64, FF=2048.  M = B*S*T = 36864.
// ---------------------------------------------------------------------------

typedef __attribute__((ext_vector_type(16))) __bf16 v16bf;
typedef __attribute__((ext_vector_type(8)))  __bf16 v8bf;
typedef __attribute__((ext_vector_type(2)))  __bf16 v2bf;
typedef __attribute__((ext_vector_type(8)))  float  v8f;

// A-matrix 16x32 bf16 fragment (ISA 7.12.2): lanes 0-15 hold K=[0..7],[16..23];
// lanes 16-31 hold K=[8..15],[24..31].  Two 16B LDS loads per lane.
__device__ __forceinline__ v16bf frag_a(const __bf16* row0, int stride, int lane) {
  const __bf16* p = row0 + (lane & 15) * stride + ((lane >> 4) << 3);
  v8bf lo = *(const v8bf*)p;
  v8bf hi = *(const v8bf*)(p + 16);
  return __builtin_shufflevector(lo, hi, 0,1,2,3,4,5,6,7,8,9,10,11,12,13,14,15);
}

// B-matrix 32x16 bf16 fragment: lanes 0-15 hold K=[0..7],[16..23]; lanes 16-31
// K=[8..15],[24..31] (same as A).  Source LDS tile stored K-major per N row (B^T).
__device__ __forceinline__ v16bf frag_b(const __bf16* row0, int stride, int lane) {
  const __bf16* p = row0 + (lane & 15) * stride + ((lane >> 4) << 3);
  v8bf lo = *(const v8bf*)p;
  v8bf hi = *(const v8bf*)(p + 16);
  return __builtin_shufflevector(lo, hi, 0,1,2,3,4,5,6,7,8,9,10,11,12,13,14,15);
}

__device__ __forceinline__ v8f wmma_bf16(v16bf a, v16bf b, v8f c) {
  return __builtin_amdgcn_wmma_f32_16x16x32_bf16(false, a, false, b, (short)0, c,
                                                 false, false);
}

// Async copy 16 bytes global->LDS (GLOBAL_LOAD_ASYNC_TO_LDS_B128, ASYNCcnt).
// Generic LDS pointer low 32 bits == workgroup-relative LDS byte address.
__device__ __forceinline__ void async_copy16(const void* gbase, unsigned byte_off,
                                             void* lds_dst) {
  asm volatile("global_load_async_to_lds_b128 %0, %1, %2 offset:0"
               :: "v"((unsigned)(uintptr_t)lds_dst), "v"(byte_off), "s"(gbase)
               : "memory");
}
__device__ __forceinline__ void async_wait0() {
  asm volatile("s_wait_asynccnt 0x0" ::: "memory");
}

// ---------------------------------------------------------------------------
// C_bf16[M,N] = act( A_bf16[M,K] @ W_f32[K,N] + bias[N] ), act = LeakyReLU opt.
// Block: 256 threads (8 waves), tile 64(M) x 256(N); each wave owns 32x64
// (2x4 sixteens -> 8 WMMAs per K-step of 32).
// ---------------------------------------------------------------------------
__global__ __launch_bounds__(256)
void gemm_kernel(const __bf16* __restrict__ A, const float* __restrict__ W,
                 const float* __restrict__ bias, __bf16* __restrict__ C,
                 unsigned M, unsigned K, unsigned N, int act) {
  __shared__ __bf16 As[64 * 32];    // [m][k]
  __shared__ __bf16 Bt[256 * 32];   // [n][k]  (K-transposed weights)
  const unsigned tid  = threadIdx.x;
  const unsigned lane = tid & 31;
  const unsigned w    = tid >> 5;
  const unsigned m0   = blockIdx.y * 64;
  const unsigned n0   = blockIdx.x * 256;
  const unsigned wm   = (w >> 2) * 32;   // 0 / 32
  const unsigned wn   = (w & 3) * 64;    // 0,64,128,192

  v8f acc[2][4] = {};

  // A-tile async chunk for this thread: 64x32 bf16 = 256 x 16B chunks.
  const unsigned arow = tid >> 2, apart = (tid & 3) * 8;
  __bf16* a_dst = &As[arow * 32 + apart];

  for (unsigned k0 = 0; k0 < K; k0 += 32) {
    // ---- stage A (raw bf16 copy, async DMA to LDS) ----
    async_copy16(A, ((m0 + arow) * K + k0 + apart) * 2u, a_dst);
    // ---- stage W (f32 -> bf16, transposed, packed pair stores) ----
#pragma unroll
    for (unsigned j = 0; j < 4; ++j) {
      unsigned u  = tid + j * 256;            // 1024 units of 2 rows x 4 cols
      unsigned kk = (u >> 6) * 2;             // even k
      unsigned nn = (u & 63) * 4;
      const float* w0 = &W[(k0 + kk) * N + n0 + nn];
      float4 r0 = *(const float4*)w0;
      float4 r1 = *(const float4*)(w0 + N);
      const float* f0 = (const float*)&r0;
      const float* f1 = (const float*)&r1;
#pragma unroll
      for (unsigned e = 0; e < 4; ++e) {
        v2bf p = { (__bf16)f0[e], (__bf16)f1[e] };
        *(v2bf*)&Bt[(nn + e) * 32 + kk] = p;
      }
    }
    async_wait0();
    __syncthreads();

    v16bf af[2], bfg[4];
#pragma unroll
    for (unsigned i = 0; i < 2; ++i) af[i]  = frag_a(&As[(wm + i * 16) * 32], 32, lane);
#pragma unroll
    for (unsigned j = 0; j < 4; ++j) bfg[j] = frag_b(&Bt[(wn + j * 16) * 32], 32, lane);
#pragma unroll
    for (unsigned i = 0; i < 2; ++i)
#pragma unroll
      for (unsigned j = 0; j < 4; ++j)
        acc[i][j] = wmma_bf16(af[i], bfg[j], acc[i][j]);
    __syncthreads();
  }

  const unsigned lh = lane >> 4, lo = lane & 15;
#pragma unroll
  for (unsigned i2 = 0; i2 < 2; ++i2) {
    unsigned row0 = m0 + wm + i2 * 16 + lh * 8;
#pragma unroll
    for (unsigned j = 0; j < 4; ++j) {
      unsigned col = n0 + wn + j * 16 + lo;
      float bv = bias[col];
#pragma unroll
      for (unsigned i = 0; i < 8; ++i) {
        float v = acc[i2][j][i] + bv;
        if (act) v = (v >= 0.f) ? v : 0.01f * v;
        C[(row0 + i) * N + col] = (__bf16)v;
      }
    }
  }
}

// ---------------------------------------------------------------------------
// Axial attention for one (b, h, row).  96 keys/queries, DK=64, bf16 q/k/v/o.
// time_mode=1: attend over T (row = s); 0: attend over S (row = t).
// ---------------------------------------------------------------------------
__global__ __launch_bounds__(256)
void attn_kernel(const __bf16* __restrict__ Qg, const __bf16* __restrict__ Kg,
                 const __bf16* __restrict__ Vg, const float* __restrict__ mask,
                 __bf16* __restrict__ Og, int time_mode) {
  __shared__ __bf16 Qs[96 * 64];   // [t][d]
  __shared__ __bf16 Ks[96 * 64];   // [u][d]
  __shared__ __bf16 Vt[64 * 96];   // [d][u]  (transposed)
  __shared__ __bf16 Pb[96 * 96];   // scores -> probabilities

  const unsigned tid  = threadIdx.x;
  const unsigned lane = tid & 31;
  const unsigned w    = tid >> 5;
  const unsigned idx  = blockIdx.x;
  const unsigned r    = idx % 96;
  const unsigned hh   = (idx / 96) & 7;
  const unsigned bb   = idx / (96 * 8);

  unsigned base, mbase, rstride, mstride;
  if (time_mode) {               // rows = t, fixed (b, s=r)
    base    = (bb * 96 + r) * 96 * 512 + hh * 64;
    rstride = 512;
    mbase   = (bb * 96 + r) * 96;        // mask[b][s][t], t = row
    mstride = 1;
  } else {                       // rows = s, fixed (b, t=r)
    base    = (bb * 96 * 96 + r) * 512 + hh * 64;
    rstride = 96 * 512;
    mbase   = bb * 96 * 96 + r;          // mask[b][s][t], s = row
    mstride = 96;
  }

  // Stage: Q,K via async DMA (raw bf16); V manually transposed.
  for (unsigned c = tid; c < 768; c += 256) {          // 96 rows x 8 chunks
    unsigned row = c >> 3, part = (c & 7) * 8;
    unsigned boff = (base + row * rstride + part) * 2u;
    async_copy16(Qg, boff, &Qs[row * 64 + part]);
    async_copy16(Kg, boff, &Ks[row * 64 + part]);
    v8bf vv = *(const v8bf*)&Vg[base + row * rstride + part];
#pragma unroll
    for (unsigned e = 0; e < 8; ++e) Vt[(part + e) * 96 + row] = vv[e];
  }
  async_wait0();
  __syncthreads();

  // scores = (Q @ K^T) / 8 ; 6x6 tiles, two K=32 WMMA steps each
  for (unsigned tile = w; tile < 36; tile += 8) {
    unsigned ti = (tile / 6) * 16, ui = (tile % 6) * 16;
    v8f acc = {};
#pragma unroll
    for (unsigned ks = 0; ks < 64; ks += 32) {
      v16bf a = frag_a(&Qs[ti * 64 + ks], 64, lane);
      v16bf b = frag_b(&Ks[ui * 64 + ks], 64, lane);
      acc = wmma_bf16(a, b, acc);
    }
    unsigned lh = lane >> 4, lo = lane & 15;
#pragma unroll
    for (unsigned i = 0; i < 8; ++i)
      Pb[(ti + lh * 8 + i) * 96 + ui + lo] = (__bf16)(acc[i] * 0.125f);
  }
  __syncthreads();

  // row softmax, post-softmax mask folded into normalizer, in place
  for (unsigned row = w; row < 96; row += 8) {
    float e0 = (float)Pb[row * 96 + lane];
    float e1 = (float)Pb[row * 96 + 32 + lane];
    float e2 = (float)Pb[row * 96 + 64 + lane];
    float mx = fmaxf(e0, fmaxf(e1, e2));
#pragma unroll
    for (int off = 16; off > 0; off >>= 1) mx = fmaxf(mx, __shfl_xor(mx, off, 32));
    e0 = __expf(e0 - mx); e1 = __expf(e1 - mx); e2 = __expf(e2 - mx);
    float s = e0 + e1 + e2;
#pragma unroll
    for (int off = 16; off > 0; off >>= 1) s += __shfl_xor(s, off, 32);
    float g = mask[mbase + row * mstride] / s;
    Pb[row * 96 + lane]      = (__bf16)(e0 * g);
    Pb[row * 96 + 32 + lane] = (__bf16)(e1 * g);
    Pb[row * 96 + 64 + lane] = (__bf16)(e2 * g);
  }
  __syncthreads();

  // O = P @ V ; 6x4 tiles, three K=32 WMMA steps each
  for (unsigned tile = w; tile < 24; tile += 8) {
    unsigned ti = (tile / 4) * 16, di = (tile % 4) * 16;
    v8f acc = {};
#pragma unroll
    for (unsigned ku = 0; ku < 96; ku += 32) {
      v16bf a = frag_a(&Pb[ti * 96 + ku], 96, lane);
      v16bf b = frag_b(&Vt[di * 96 + ku], 96, lane);
      acc = wmma_bf16(a, b, acc);
    }
    unsigned lh = lane >> 4, lo = lane & 15;
    unsigned row0 = ti + lh * 8, col = di + lo;
#pragma unroll
    for (unsigned i = 0; i < 8; ++i)
      Og[base + (row0 + i) * rstride + col] = (__bf16)acc[i];
  }
}

// ---------------------------------------------------------------------------
// out = LayerNorm(a + b) * gamma + beta.  One 256-thread block per token.
// ---------------------------------------------------------------------------
template <bool A_F32, bool OUT_F32>
__global__ __launch_bounds__(256)
void ln_kernel(const void* __restrict__ Ap, const __bf16* __restrict__ Bres,
               const float* __restrict__ gamma, const float* __restrict__ beta,
               void* __restrict__ outp) {
  __shared__ float red[16];
  const unsigned base = blockIdx.x * 512;
  const unsigned tid = threadIdx.x, lane = tid & 31, w = tid >> 5;
  float a0 = A_F32 ? ((const float*)Ap)[base + tid]
                   : (float)((const __bf16*)Ap)[base + tid];
  float a1 = A_F32 ? ((const float*)Ap)[base + tid + 256]
                   : (float)((const __bf16*)Ap)[base + tid + 256];
  float y0 = a0 + (float)Bres[base + tid];
  float y1 = a1 + (float)Bres[base + tid + 256];
  float s = y0 + y1;
  float q = y0 * y0 + y1 * y1;
#pragma unroll
  for (int off = 16; off > 0; off >>= 1) {
    s += __shfl_xor(s, off, 32);
    q += __shfl_xor(q, off, 32);
  }
  if (lane == 0) { red[w] = s; red[8 + w] = q; }
  __syncthreads();
  float S = 0.f, Q = 0.f;
#pragma unroll
  for (int i = 0; i < 8; ++i) { S += red[i]; Q += red[8 + i]; }
  const float mean = S * (1.f / 512.f);
  const float var  = Q * (1.f / 512.f) - mean * mean;
  const float inv  = rsqrtf(var + 1e-5f);
  float o0 = (y0 - mean) * inv * gamma[tid] + beta[tid];
  float o1 = (y1 - mean) * inv * gamma[tid + 256] + beta[tid + 256];
  if (OUT_F32) {
    ((float*)outp)[base + tid] = o0;
    ((float*)outp)[base + tid + 256] = o1;
  } else {
    ((__bf16*)outp)[base + tid] = (__bf16)o0;
    ((__bf16*)outp)[base + tid + 256] = (__bf16)o1;
  }
}

// one-shot f32 -> bf16 conversion (x only), 8 elems/thread
__global__ __launch_bounds__(256)
void cvt_kernel(const float* __restrict__ in, __bf16* __restrict__ out) {
  unsigned i = (blockIdx.x * 256 + threadIdx.x) * 8;
  float4 a = *(const float4*)&in[i];
  float4 b = *(const float4*)&in[i + 4];
  v8bf o = { (__bf16)a.x, (__bf16)a.y, (__bf16)a.z, (__bf16)a.w,
             (__bf16)b.x, (__bf16)b.y, (__bf16)b.z, (__bf16)b.w };
  *(v8bf*)&out[i] = o;
}

// ---------------------------------------------------------------------------
extern "C" void kernel_launch(void* const* d_in, const int* in_sizes, int n_in,
                              void* d_out, int out_size, void* d_ws, size_t ws_size,
                              hipStream_t stream) {
  (void)in_sizes; (void)n_in; (void)out_size; (void)ws_size;
  const float* x     = (const float*)d_in[0];
  const float* mask  = (const float*)d_in[1];
  const float* t_w   = (const float*)d_in[2];
  const float* t_b   = (const float*)d_in[3];
  const float* s_w   = (const float*)d_in[4];
  const float* s_b   = (const float*)d_in[5];
  const float* fc_w1 = (const float*)d_in[6];
  const float* fc_b1 = (const float*)d_in[7];
  const float* fc_w2 = (const float*)d_in[8];
  const float* fc_b2 = (const float*)d_in[9];
  const float* ln1_g = (const float*)d_in[10];
  const float* ln1_b = (const float*)d_in[11];
  const float* ln2_g = (const float*)d_in[12];
  const float* ln2_b = (const float*)d_in[13];

  const unsigned M = 4 * 96 * 96;      // 36864 tokens
  const unsigned D = 512, FF = 2048;
  const size_t ND = (size_t)M * D;
  const size_t DD = (size_t)D * D;

  __bf16* wsb = (__bf16*)d_ws;
  __bf16* xbf = wsb + 0 * ND;
  __bf16* q   = wsb + 1 * ND;
  __bf16* k   = wsb + 2 * ND;
  __bf16* v   = wsb + 3 * ND;
  __bf16* ctx = wsb + 4 * ND;
  __bf16* tt  = wsb + 5 * ND;
  __bf16* ss  = wsb + 6 * ND;
  __bf16* x1  = wsb + 7 * ND;
  __bf16* ff  = wsb + 8 * ND;
  __bf16* h   = wsb + 1 * ND;          // reuses q/k/v/ctx region (4*ND), dead by FFN

  const dim3 blk(256);
  auto gemm = [&](const __bf16* A, const float* W, const float* bias, __bf16* C,
                  unsigned mm, unsigned kk, unsigned nn, int act) {
    dim3 grid(nn / 256, mm / 64);
    gemm_kernel<<<grid, blk, 0, stream>>>(A, W, bias, C, mm, kk, nn, act);
  };
  const int attn_blocks = 4 * 8 * 96;  // B * H * 96

  cvt_kernel<<<ND / (256 * 8), blk, 0, stream>>>(x, xbf);

  // ---- time-axis MHA -------------------------------------------------------
  gemm(xbf, t_w + 0 * DD, t_b + 0 * D, q, M, D, D, 0);
  gemm(xbf, t_w + 1 * DD, t_b + 1 * D, k, M, D, D, 0);
  gemm(xbf, t_w + 2 * DD, t_b + 2 * D, v, M, D, D, 0);
  attn_kernel<<<attn_blocks, blk, 0, stream>>>(q, k, v, mask, ctx, 1);
  gemm(ctx, t_w + 3 * DD, t_b + 3 * D, tt, M, D, D, 0);

  // ---- space-axis MHA ------------------------------------------------------
  gemm(tt, s_w + 0 * DD, s_b + 0 * D, q, M, D, D, 0);
  gemm(tt, s_w + 1 * DD, s_b + 1 * D, k, M, D, D, 0);
  gemm(tt, s_w + 2 * DD, s_b + 2 * D, v, M, D, D, 0);
  attn_kernel<<<attn_blocks, blk, 0, stream>>>(q, k, v, mask, ctx, 0);
  gemm(ctx, s_w + 3 * DD, s_b + 3 * D, ss, M, D, D, 0);

  // ---- LN1, FFN, LN2 -------------------------------------------------------
  ln_kernel<true, false><<<M, blk, 0, stream>>>(x, ss, ln1_g, ln1_b, x1);
  gemm(x1, fc_w1, fc_b1, h, M, D, FF, 1);     // + LeakyReLU
  gemm(h, fc_w2, fc_b2, ff, M, FF, D, 0);
  ln_kernel<false, true><<<M, blk, 0, stream>>>(x1, ff, ln2_g, ln2_b, d_out);
}
